// SOM2D_3375844294845
// MI455X (gfx1250) — matrix operand order repaired and checked by reading, not compiled, not run on hardware
//
#include <hip/hip_runtime.h>

// ---------------- problem constants ----------------
#define NS 32768              // samples
#define MU 4096               // SOM units (64x64)
#define DD 128                // input dim
#define KCH (DD / 32)         // 4 K-chunks of 32 per D
#define NTILE_M (MU / 16)     // 256 unit tiles
#define NSTRIP (NS / 16)      // 2048 sample strips
#define WAVES_PER_BLOCK 8     // 256 threads, wave32

#define FRAG_BYTES 1024       // 32 lanes * 32 B per bf16 fragment
#define TILE_BYTES (8 * FRAG_BYTES)  // 4 chunks x {hi,lo} = 8 KB per unit tile

typedef __attribute__((ext_vector_type(16))) __bf16 v16bf;
typedef __attribute__((ext_vector_type(8)))  float  v8f;

// ---------------- workspace layout (bytes) ----------------
#define SZ_A   ((size_t)NS * DD * 2)   // 8 MB each (hi, lo)
#define SZ_B   ((size_t)MU * DD * 2)   // 1 MB each (hi, lo)
#define OFF_AHI ((size_t)0)
#define OFF_ALO (OFF_AHI + SZ_A)
#define OFF_BHI (OFF_ALO + SZ_A)
#define OFF_BLO (OFF_BHI + SZ_B)
#define OFF_WSQ (OFF_BLO + SZ_B)       // MU floats

// =====================================================================
// CDNA5 async copy: global -> LDS, 16 B per lane, tracked by ASYNCcnt.
// Inline asm (portable across ROCm7.2 / amdgpu-toolchain builtin arity).
// =====================================================================
__device__ __forceinline__ void async_b128(unsigned int ldsoff, const void* gaddr) {
    asm volatile("global_load_async_to_lds_b128 %0, %1, off"
                 :: "v"(ldsoff), "v"(gaddr) : "memory");
}
__device__ __forceinline__ void wait_async0() {
    asm volatile("s_wait_asynccnt 0x0" ::: "memory");
}

// =====================================================================
// Prep: split f32 -> bf16 hi/lo and pre-swizzle into WMMA fragment order.
// A-fragment (16x32 bf16): lane L holds row M = L%16.
//   lanes 0-15 : elem e -> K = e + (e>=8 ? 8 : 0)
//   lanes 16-31: elem e -> K = e + (e>=8 ? 8 : 0) + 8
// B-fragment (32x16 bf16): lane L holds col N = L%16; elem e -> K = e + (e>=8 ? 8 : 0) + 8*(L>=16)
// Destination is fragment-major == flat thread id.
// =====================================================================
__global__ void som_prep_A(const float* __restrict__ x,
                           __bf16* __restrict__ ahi, __bf16* __restrict__ alo) {
    int t = blockIdx.x * blockDim.x + threadIdx.x;
    if (t >= NS * DD) return;
    int e     = t & 15;
    int lane  = (t >> 4) & 31;
    int chunk = (t >> 9) & (KCH - 1);
    int strip = t >> 11;
    int lg    = lane >> 4;
    int row   = strip * 16 + (lane & 15);
    int kl    = e + ((e >= 8) ? 8 : 0) + (lg ? 8 : 0);
    float w   = x[row * DD + chunk * 32 + kl];
    __bf16 hi = (__bf16)w;
    __bf16 lo = (__bf16)(w - (float)hi);
    ahi[t] = hi;
    alo[t] = lo;
}

__global__ void som_prep_B(const float* __restrict__ wt,
                           __bf16* __restrict__ bhi, __bf16* __restrict__ blo) {
    int t = blockIdx.x * blockDim.x + threadIdx.x;
    if (t >= MU * DD) return;
    int e     = t & 15;
    int lane  = (t >> 4) & 31;
    int chunk = (t >> 9) & (KCH - 1);
    int tile  = t >> 11;
    int lg    = lane >> 4;
    int col   = tile * 16 + (lane & 15);
    int kl    = e + ((e >= 8) ? 8 : 0) + (lg ? 8 : 0);
    float w   = wt[col * DD + chunk * 32 + kl];
    __bf16 hi = (__bf16)w;
    __bf16 lo = (__bf16)(w - (float)hi);
    bhi[t] = hi;
    blo[t] = lo;
}

__global__ void som_wsq(const float* __restrict__ wt, float* __restrict__ wsq) {
    int m = blockIdx.x * blockDim.x + threadIdx.x;
    if (m >= MU) return;
    float s = 0.f;
#pragma unroll 8
    for (int d = 0; d < DD; ++d) {
        float v = wt[m * DD + d];
        s += v * v;
    }
    wsq[m] = s;
}

// =====================================================================
// Main: block = 8 waves; each wave owns a 16-row sample strip.
// B tiles (8 KB: 4 chunks x hi/lo) are staged into double-buffered LDS by
// async global->LDS copies shared by all 8 waves (8x traffic reduction),
// overlapped with the 12 WMMAs of the current tile.
// cross ~= hi*hi + hi*lo + lo*hi  (bf16x3 split => ~f32 accuracy)
// score = ||w||^2 - 2*cross  (x^2 term dropped: constant per row)
// =====================================================================
__global__ __launch_bounds__(256) void som_wmma_argmin(
    const __bf16* __restrict__ ahi, const __bf16* __restrict__ alo,
    const __bf16* __restrict__ bhi, const __bf16* __restrict__ blo,
    const float* __restrict__ wsq, const int* __restrict__ grid2,
    int* __restrict__ out) {
    __shared__ __align__(32) char smem[2 * TILE_BYTES];  // 16 KB double buffer

    const int u     = threadIdx.x;
    const int lane  = u & 31;
    const int wave  = u >> 5;
    const int strip = blockIdx.x * WAVES_PER_BLOCK + wave;
    const int colL  = lane & 15;

    const char* gBH = (const char*)bhi;
    const char* gBL = (const char*)blo;

    // ---- stage helper: 512 x 16B segments per tile, 2 per thread ----
    auto stage_tile = [&](int t, int buf) {
#pragma unroll
        for (int k = 0; k < 2; ++k) {
            int s   = u + k * 256;        // 0..511
            int f   = s >> 6;             // fragment 0..7 (c*2 + hi/lo)
            int off = (s & 63) * 16;      // byte offset within fragment
            int c   = f >> 1;
            int h   = f & 1;
            const char* src = (h ? gBL : gBH) +
                              (size_t)t * (KCH * FRAG_BYTES) + c * FRAG_BYTES + off;
            unsigned int dst = (unsigned int)(uintptr_t)(
                smem + buf * TILE_BYTES + f * FRAG_BYTES + off);
            async_b128(dst, src);
        }
    };

    // ---- load the wave's 16x128 A fragments (hi/lo), VGPR-resident ----
    const v16bf* pAH = (const v16bf*)ahi;
    const v16bf* pAL = (const v16bf*)alo;
    v16bf AH[KCH], AL[KCH];
#pragma unroll
    for (int c = 0; c < KCH; ++c) {
        size_t fi = ((size_t)strip * KCH + c) * 32 + lane;
        AH[c] = pAH[fi];
        AL[c] = pAL[fi];
    }

    float minv[8];
    int   mini[8];
#pragma unroll
    for (int r = 0; r < 8; ++r) { minv[r] = __builtin_inff(); mini[r] = 0; }

    // ---- prologue: stage tile 0 ----
    stage_tile(0, 0);
    wait_async0();
    __syncthreads();

    for (int t = 0; t < NTILE_M; ++t) {
        const int buf = t & 1;
        if (t + 1 < NTILE_M) stage_tile(t + 1, buf ^ 1);  // overlap with compute

        const char* base = smem + buf * TILE_BYTES + lane * 32;
        v8f acc = {};
#pragma unroll
        for (int c = 0; c < KCH; ++c) {
            v16bf BH = *(const v16bf*)(base + (c * 2 + 0) * FRAG_BYTES);
            v16bf BL = *(const v16bf*)(base + (c * 2 + 1) * FRAG_BYTES);
            acc = __builtin_amdgcn_wmma_f32_16x16x32_bf16(
                false, AH[c], false, BH, (short)0, acc, false, false);
            acc = __builtin_amdgcn_wmma_f32_16x16x32_bf16(
                false, AH[c], false, BL, (short)0, acc, false, false);
            acc = __builtin_amdgcn_wmma_f32_16x16x32_bf16(
                false, AL[c], false, BH, (short)0, acc, false, false);
        }

        const int   mcol = t * 16 + colL;
        const float q    = wsq[mcol];
#pragma unroll
        for (int r = 0; r < 8; ++r) {
            float s = __builtin_fmaf(-2.0f, acc[r], q);
            // per-lane m index strictly increases with t -> strict '<'
            // preserves argmin first-occurrence semantics within a lane.
            if (s < minv[r]) { minv[r] = s; mini[r] = mcol; }
        }

        wait_async0();     // next tile landed in the alternate buffer
        __syncthreads();   // all waves done reading buf before it is reused
    }

    // Reduce across the 16 lanes of each column group (xor masks 1..8 stay
    // within lanes [0,16) / [16,32) on wave32). Tie-break: lower unit index.
#pragma unroll
    for (int m = 1; m < 16; m <<= 1) {
#pragma unroll
        for (int r = 0; r < 8; ++r) {
            float ov = __shfl_xor(minv[r], m, 32);
            int   oi = __shfl_xor(mini[r], m, 32);
            if (ov < minv[r] || (ov == minv[r] && oi < mini[r])) {
                minv[r] = ov;
                mini[r] = oi;
            }
        }
    }

    // Lane 0 owns rows 0..7, lane 16 owns rows 8..15.
    if (colL == 0) {
        int rowbase = strip * 16 + (lane >> 4) * 8;
#pragma unroll
        for (int r = 0; r < 8; ++r) {
            int idx = mini[r];
            out[(rowbase + r) * 2 + 0] = grid2[idx * 2 + 0];
            out[(rowbase + r) * 2 + 1] = grid2[idx * 2 + 1];
        }
    }
}

// =====================================================================
extern "C" void kernel_launch(void* const* d_in, const int* in_sizes, int n_in,
                              void* d_out, int out_size, void* d_ws, size_t ws_size,
                              hipStream_t stream) {
    const float* inputs  = (const float*)d_in[0];  // [NS, DD] f32
    const float* weights = (const float*)d_in[1];  // [MU, DD] f32
    const int*   grid2   = (const int*)d_in[2];    // [MU, 2]  i32
    int*         out     = (int*)d_out;            // [NS, 2]  i32

    char* ws = (char*)d_ws;
    __bf16* ahi = (__bf16*)(ws + OFF_AHI);
    __bf16* alo = (__bf16*)(ws + OFF_ALO);
    __bf16* bhi = (__bf16*)(ws + OFF_BHI);
    __bf16* blo = (__bf16*)(ws + OFF_BLO);
    float*  wsq = (float*)(ws + OFF_WSQ);

    som_prep_A<<<(NS * DD) / 256, 256, 0, stream>>>(inputs, ahi, alo);
    som_prep_B<<<(MU * DD) / 256, 256, 0, stream>>>(weights, bhi, blo);
    som_wsq<<<MU / 256, 256, 0, stream>>>(weights, wsq);

    som_wmma_argmin<<<NSTRIP / WAVES_PER_BLOCK, 32 * WAVES_PER_BLOCK, 0, stream>>>(
        ahi, alo, bhi, blo, wsq, grid2, out);
}
